// bipartite_msg_36773509988939
// MI455X (gfx1250) — matrix operand order repaired and checked by reading, not compiled, not run on hardware
//
#include <hip/hip_runtime.h>

#define EPS_BN 1e-5f
#define SLOPE  0.01f

#define ROWS   16384
#define DIM_D  256
#define DIM_2D 512
#define DIM_H  1024

typedef __attribute__((ext_vector_type(16))) __bf16 v16bf;
typedef __attribute__((ext_vector_type(8)))  __bf16 v8bf;
typedef __attribute__((ext_vector_type(8)))  float  v8f;
typedef __attribute__((ext_vector_type(4)))  float  v4f;

union Frag16 { v16bf v; v8bf h[2]; };

// ---------------------------------------------------------------------------
// Fused bf16 WMMA GEMM:  C[m,n] = sum_k A_f32[m,k] * B_bf16[k,n]  (+epilogue)
// Block = 256 threads = 8 waves. Block tile: 64 rows x (NACC*32) cols.
// Wave w: m-strip (w>>1)*16, n-half (w&1)*(NACC*16); NACC accumulators.
// ---------------------------------------------------------------------------
enum { MODE_STORE_BF16 = 0, MODE_STORE_F32 = 1, MODE_LEAKY = 2, MODE_LEAKY_RES = 3 };

#define TM 64
#define TK 32
#define LDS_A 40  // 32 + 8 pad (bf16 elems)

template <int MODE, int NACC>
__global__ __launch_bounds__(256) void gemm_bf16_wmma(
    const float* __restrict__ A, int lda,
    const __bf16* __restrict__ B, int ldb,
    const float* __restrict__ bias,
    const float* __restrict__ res, int ldres,
    float* __restrict__ outF, __bf16* __restrict__ outB,
    int ldc, int outColOff, int K)
{
    constexpr int TN    = NACC * 32;   // block tile width
    constexpr int LDS_B = TN + 8;      // padded row (bf16 elems), keeps 16B align

    __shared__ __bf16 shA[2][TM * LDS_A];
    __shared__ __bf16 shB[2][TK * LDS_B];

    const int tid  = threadIdx.x;
    const int lane = tid & 31;
    const int wave = tid >> 5;
    const int l16  = lane & 15;
    const int lHi  = lane >> 4;
    const int mSub = (wave >> 1) * 16;
    const int nW   = (wave & 1) * (NACC * 16);

    const int mBase = blockIdx.y * TM;
    const int nBase = blockIdx.x * TN;

    // A staging: 64x32 f32 -> bf16, 8 contiguous elems / thread
    const int aRow = tid >> 2;
    const int aCol = (tid & 3) * 8;
    // B staging: 32xTN bf16, TN/64 x v8bf / thread
    const int bRow = tid >> 3;
    const int bCol = (tid & 7) * 8;

    v8f acc[NACC];
#pragma unroll
    for (int j = 0; j < NACC; ++j) acc[j] = (v8f){0.f,0.f,0.f,0.f,0.f,0.f,0.f,0.f};

    auto loadTiles = [&](int buf, int kt) {
        const int k0 = kt * TK;
        const float* ap = A + (size_t)(mBase + aRow) * (size_t)lda + (size_t)(k0 + aCol);
        v4f a0 = *(const v4f*)ap;
        v4f a1 = *(const v4f*)(ap + 4);
        __bf16* as = &shA[buf][aRow * LDS_A + aCol];
        as[0]=(__bf16)a0.x; as[1]=(__bf16)a0.y; as[2]=(__bf16)a0.z; as[3]=(__bf16)a0.w;
        as[4]=(__bf16)a1.x; as[5]=(__bf16)a1.y; as[6]=(__bf16)a1.z; as[7]=(__bf16)a1.w;
#pragma unroll
        for (int j = 0; j < TN / 64; ++j) {
            const __bf16* bp = B + (size_t)(k0 + bRow) * (size_t)ldb
                                 + (size_t)(nBase + bCol + j * 64);
            *(v8bf*)&shB[buf][bRow * LDS_B + bCol + j * 64] = *(const v8bf*)bp;
        }
    };

    const int KT = K / TK;
    loadTiles(0, 0);
    __syncthreads();

    for (int kt = 0; kt < KT; ++kt) {
        const int buf = kt & 1;
        if (kt + 1 < KT) loadTiles(buf ^ 1, kt + 1);

        // A fragment: lane holds row (mSub+l16); lanes<16: K 0-7,16-23; lanes>=16: K 8-15,24-31
        Frag16 a;
        const __bf16* arow = &shA[buf][(mSub + l16) * LDS_A + lHi * 8];
        a.h[0] = *(const v8bf*)(arow);
        a.h[1] = *(const v8bf*)(arow + 16);

#pragma unroll
        for (int j = 0; j < NACC; ++j) {
            // B fragment: lane holds col (l16); lanes<16: K 0-7,16-23; lanes>=16: K 8-15,24-31
            Frag16 b;
            const __bf16* brow = &shB[buf][(lHi * 8) * LDS_B + nW + j * 16 + l16];
#pragma unroll
            for (int s = 0; s < 8; ++s) {
                b.v[s]     = brow[s * LDS_B];
                b.v[8 + s] = brow[(16 + s) * LDS_B];
            }
            acc[j] = __builtin_amdgcn_wmma_f32_16x16x32_bf16(
                false, a.v, false, b.v, (short)0, acc[j], false, false);
        }
        __syncthreads();
    }

    // Epilogue: C layout -> lane l, vgpr i : (m = mSub + lHi*8 + i, n = nLocal + l16)
    const int row0 = mBase + mSub + lHi * 8;
#pragma unroll
    for (int j = 0; j < NACC; ++j) {
        const int col = nBase + nW + j * 16 + l16;
#pragma unroll
        for (int i = 0; i < 8; ++i) {
            float v = acc[j][i];
            const int r = row0 + i;
            if (MODE != MODE_STORE_F32) v += bias[col];
            if (MODE == MODE_LEAKY || MODE == MODE_LEAKY_RES)
                v = (v >= 0.f) ? v : SLOPE * v;
            if (MODE == MODE_LEAKY_RES)
                v += res[(size_t)r * ldres + col];
            if (MODE == MODE_STORE_BF16)
                outB[(size_t)r * ldc + col] = (__bf16)v;
            else
                outF[(size_t)r * ldc + (outColOff + col)] = v;
        }
    }
}

// ---------------------------------------------------------------------------
// Helper kernels
// ---------------------------------------------------------------------------

// xcat[:, 0:256] = N
__global__ void copy_n_kernel(const float* __restrict__ Nin, float* __restrict__ xcat) {
    int idx = blockIdx.x * blockDim.x + threadIdx.x;   // < ROWS*DIM_D
    int r = idx >> 8, c = idx & 255;
    xcat[(size_t)r * DIM_2D + c] = Nin[idx];
}

// Deterministic two-pass column stats: partial sums per (chunk, col)
__global__ void col_stats_partial(const float* __restrict__ X, int ld, int ncols,
                                  int rowsPerChunk,
                                  float* __restrict__ psum, float* __restrict__ psq) {
    int col = blockIdx.x * blockDim.x + threadIdx.x;
    int r0  = blockIdx.y * rowsPerChunk;
    const float* p = X + (size_t)r0 * ld + col;
    float s = 0.f, q = 0.f;
    for (int r = 0; r < rowsPerChunk; ++r) { float v = p[(size_t)r * ld]; s += v; q += v * v; }
    psum[blockIdx.y * ncols + col] = s;
    psq [blockIdx.y * ncols + col] = q;
}

// s = gamma * rsqrt(var+eps), t = beta - mu*s   (BN folded as x*s + t)
__global__ void col_stats_finalize(const float* __restrict__ psum, const float* __restrict__ psq,
                                   int nchunks, int ncols, float invRows,
                                   const float* __restrict__ gamma, const float* __restrict__ beta,
                                   float* __restrict__ s, float* __restrict__ t) {
    int col = blockIdx.x * blockDim.x + threadIdx.x;
    if (col >= ncols) return;
    float su = 0.f, sq = 0.f;
    for (int c = 0; c < nchunks; ++c) { su += psum[c * ncols + col]; sq += psq[c * ncols + col]; }
    float mu  = su * invRows;
    float var = fmaxf(sq * invRows - mu * mu, 0.f);
    float sc  = gamma[col] * rsqrtf(var + EPS_BN);
    s[col] = sc;
    t[col] = beta[col] - mu * sc;
}

// Wt[c,n] = bf16( W[n,c] * (s?s[c]:1) )   (transpose + BN scale fold)
__global__ void fold_weight_kernel(const float* __restrict__ W, const float* __restrict__ s,
                                   __bf16* __restrict__ Wt, int C, int NN) {
    int idx = blockIdx.x * blockDim.x + threadIdx.x;
    if (idx >= C * NN) return;
    int c = idx / NN, n = idx - c * NN;
    float w = W[(size_t)n * C + c];
    if (s) w *= s[c];
    Wt[idx] = (__bf16)w;
}

// bout[n] = b[n] + sum_c t[c]*W[n,c]      (BN shift fold into bias)
__global__ void fold_bias_kernel(const float* __restrict__ W, const float* __restrict__ b,
                                 const float* __restrict__ t, float* __restrict__ bout,
                                 int C, int NN) {
    int n = blockIdx.x * blockDim.x + threadIdx.x;
    if (n >= NN) return;
    float acc = b[n];
    const float* wr = W + (size_t)n * C;
    for (int c = 0; c < C; ++c) acc += t[c] * wr[c];
    bout[n] = acc;
}

// ---------------------------------------------------------------------------
// Workspace layout (bytes)
// ---------------------------------------------------------------------------
constexpr size_t OFF_MH    = 0;                                    // 16384x256 bf16
constexpr size_t OFF_XCAT  = OFF_MH    + (size_t)ROWS * DIM_D * 2; // 16384x512 f32
constexpr size_t OFF_H1    = OFF_XCAT  + (size_t)ROWS * DIM_2D * 4;// 16384x1024 f32
constexpr size_t OFF_WINT  = OFF_H1    + (size_t)ROWS * DIM_H * 4; // 256x256 bf16
constexpr size_t OFF_W1T   = OFF_WINT  + (size_t)DIM_D * DIM_D * 2;   // 512x1024 bf16
constexpr size_t OFF_W2T   = OFF_W1T   + (size_t)DIM_2D * DIM_H * 2;  // 1024x256 bf16
constexpr size_t OFF_B1F   = OFF_W2T   + (size_t)DIM_H * DIM_D * 2;   // 1024 f32
constexpr size_t OFF_B2F   = OFF_B1F   + DIM_H * 4;                   // 256 f32
constexpr size_t OFF_PSUM1 = OFF_B2F   + DIM_D * 4;                   // 32x512 f32
constexpr size_t OFF_PSQ1  = OFF_PSUM1 + 32 * DIM_2D * 4;
constexpr size_t OFF_PSUM2 = OFF_PSQ1  + 32 * DIM_2D * 4;             // 32x1024 f32
constexpr size_t OFF_PSQ2  = OFF_PSUM2 + 32 * DIM_H * 4;
constexpr size_t OFF_S1    = OFF_PSQ2  + 32 * DIM_H * 4;
constexpr size_t OFF_T1    = OFF_S1 + DIM_2D * 4;
constexpr size_t OFF_S2    = OFF_T1 + DIM_2D * 4;
constexpr size_t OFF_T2    = OFF_S2 + DIM_H * 4;

extern "C" void kernel_launch(void* const* d_in, const int* in_sizes, int n_in,
                              void* d_out, int out_size, void* d_ws, size_t ws_size,
                              hipStream_t stream) {
    (void)in_sizes; (void)n_in; (void)out_size; (void)ws_size;
    const float* M    = (const float*)d_in[0];
    const float* Nin  = (const float*)d_in[1];
    const float* adj  = (const float*)d_in[2];
    const float* W_in = (const float*)d_in[3];
    const float* b_in = (const float*)d_in[4];
    const float* g1   = (const float*)d_in[5];
    const float* be1  = (const float*)d_in[6];
    const float* W1   = (const float*)d_in[7];
    const float* b1   = (const float*)d_in[8];
    const float* g2   = (const float*)d_in[9];
    const float* be2  = (const float*)d_in[10];
    const float* W2   = (const float*)d_in[11];
    const float* b2   = (const float*)d_in[12];
    float* outF = (float*)d_out;

    char* ws = (char*)d_ws;
    __bf16* Mh   = (__bf16*)(ws + OFF_MH);
    float*  xcat = (float*) (ws + OFF_XCAT);
    float*  h1   = (float*) (ws + OFF_H1);
    __bf16* Wint = (__bf16*)(ws + OFF_WINT);
    __bf16* W1t  = (__bf16*)(ws + OFF_W1T);
    __bf16* W2t  = (__bf16*)(ws + OFF_W2T);
    float*  b1f  = (float*) (ws + OFF_B1F);
    float*  b2f  = (float*) (ws + OFF_B2F);
    float*  ps1  = (float*) (ws + OFF_PSUM1);
    float*  pq1  = (float*) (ws + OFF_PSQ1);
    float*  ps2  = (float*) (ws + OFF_PSUM2);
    float*  pq2  = (float*) (ws + OFF_PSQ2);
    float*  s1   = (float*) (ws + OFF_S1);
    float*  t1   = (float*) (ws + OFF_T1);
    float*  s2   = (float*) (ws + OFF_S2);
    float*  t2   = (float*) (ws + OFF_T2);

    const float invRows = 1.0f / (float)ROWS;

    // 1) Wint[k,n] = bf16(W_in[n,k])
    fold_weight_kernel<<<(DIM_D * DIM_D + 255) / 256, 256, 0, stream>>>(
        W_in, nullptr, Wint, DIM_D, DIM_D);

    // 2) Mh = M @ W_inT + b_in  -> bf16 [16384,256]
    gemm_bf16_wmma<MODE_STORE_BF16, 2><<<dim3(DIM_D / 64, ROWS / 64), 256, 0, stream>>>(
        M, DIM_D, Wint, DIM_D, b_in, nullptr, 0, nullptr, Mh, DIM_D, 0, DIM_D);

    // 3) xcat[:,0:256] = N
    copy_n_kernel<<<(ROWS * DIM_D) / 256, 256, 0, stream>>>(Nin, xcat);

    // 4) xcat[:,256:512] = adj @ Mh   (full-width blocks: adj read exactly once, ~46us floor)
    gemm_bf16_wmma<MODE_STORE_F32, 8><<<dim3(1, ROWS / 64), 256, 0, stream>>>(
        adj, ROWS, Mh, DIM_D, nullptr, nullptr, 0, xcat, nullptr, DIM_2D, DIM_D, ROWS);

    // 5) BN1 stats over xcat -> fold into (W1, b1)
    col_stats_partial<<<dim3(DIM_2D / 256, 32), 256, 0, stream>>>(
        xcat, DIM_2D, DIM_2D, ROWS / 32, ps1, pq1);
    col_stats_finalize<<<(DIM_2D + 255) / 256, 256, 0, stream>>>(
        ps1, pq1, 32, DIM_2D, invRows, g1, be1, s1, t1);
    fold_weight_kernel<<<(DIM_2D * DIM_H + 255) / 256, 256, 0, stream>>>(
        W1, s1, W1t, DIM_2D, DIM_H);
    fold_bias_kernel<<<(DIM_H + 255) / 256, 256, 0, stream>>>(
        W1, b1, t1, b1f, DIM_2D, DIM_H);

    // 6) h1 = leaky(BN1(xcat) @ W1T + b1)
    gemm_bf16_wmma<MODE_LEAKY, 2><<<dim3(DIM_H / 64, ROWS / 64), 256, 0, stream>>>(
        xcat, DIM_2D, W1t, DIM_H, b1f, nullptr, 0, h1, nullptr, DIM_H, 0, DIM_2D);

    // 7) BN2 stats over h1 -> fold into (W2, b2)
    col_stats_partial<<<dim3(DIM_H / 256, 32), 256, 0, stream>>>(
        h1, DIM_H, DIM_H, ROWS / 32, ps2, pq2);
    col_stats_finalize<<<(DIM_H + 255) / 256, 256, 0, stream>>>(
        ps2, pq2, 32, DIM_H, invRows, g2, be2, s2, t2);
    fold_weight_kernel<<<(DIM_H * DIM_D + 255) / 256, 256, 0, stream>>>(
        W2, s2, W2t, DIM_H, DIM_D);
    fold_bias_kernel<<<(DIM_D + 255) / 256, 256, 0, stream>>>(
        W2, b2, t2, b2f, DIM_H, DIM_D);

    // 8) out = N + leaky(BN2(h1) @ W2T + b2)
    gemm_bf16_wmma<MODE_LEAKY_RES, 2><<<dim3(DIM_D / 64, ROWS / 64), 256, 0, stream>>>(
        h1, DIM_H, W2t, DIM_D, b2f, Nin, DIM_D, outF, nullptr, DIM_D, 0, DIM_H);
}
